// NetVLAD_24953759989818
// MI455X (gfx1250) — matrix operand order repaired and checked by reading, not compiled, not run on hardware
//
#include <hip/hip_runtime.h>

typedef __attribute__((ext_vector_type(16))) _Float16 v16h;
typedef __attribute__((ext_vector_type(4)))  _Float16 h4;
typedef __attribute__((ext_vector_type(8)))  float    v8f;

#define N_IMG   32
#define K_CL    64
#define C_DIM   128
#define P_TOT   4096
#define WG_PER_N 16
#define PBLOCK  (P_TOT / WG_PER_N)   // 256 positions per WG
#define PT      32                    // positions per iteration
#define ITERS   (PBLOCK / PT)         // 8

// LDS row strides (elements). Even (4B pair alignment) and 8B-divisible rows.
#define XN_STR   36    // xn [C][PT]  (c-major, p contiguous)  f16   row = 72 B
#define XNT_STR  132   // xnT[PT][C]  (p-major, c contiguous)  f16   row = 264 B
#define LG_STR   33    // logits f32
#define A_STR    36    // soft-assign f16

__global__ __launch_bounds__(256) void netvlad_zero(float* p, int n) {
    int i = blockIdx.x * 256 + threadIdx.x;
    if (i < n) p[i] = 0.0f;
}

__global__ __launch_bounds__(256) void netvlad_main(const float* __restrict__ x,
                                                    const float* __restrict__ conv_w,
                                                    const float* __restrict__ conv_b,
                                                    float* __restrict__ acc,
                                                    float* __restrict__ asum) {
    const int n  = blockIdx.x;       // image
    const int pb = blockIdx.y;       // position block
    const int t  = threadIdx.x;
    const int lane = t & 31;
    const int w  = t >> 5;           // wave 0..7
    const int hi = (lane >> 4) & 1;  // lane half
    const int lm = lane & 15;

    __shared__ _Float16 xn [C_DIM][XN_STR];
    __shared__ _Float16 xnT[PT][XNT_STR];
    __shared__ float    lg [K_CL][LG_STR];
    __shared__ _Float16 aa [K_CL][A_STR];
    __shared__ float    red[32][33];   // normalize: [cgrp][p]; softmax: [seg][p]
    __shared__ float    rowm[32];
    __shared__ float    rowinv[32];

    const int ktile = w & 3;   // which 16-row cluster tile this wave owns
    const int cgrp  = w >> 2;  // 0/1: position-group (GEMM1) / channel-group (GEMM2)

    // ---- Preload conv_w A-fragments (16x32 f16, contiguous k = 16*hi + s layout) for this ktile ----
    v16h aw[4];
#pragma unroll
    for (int cs = 0; cs < 4; ++cs) {
        union { v16h h; unsigned int u[8]; } fr;
        const int row = ktile * 16 + lm;
        const int b1  = hi ? 16 : 0;
#pragma unroll
        for (int v = 0; v < 8; ++v) {
            const int c = cs * 32 + b1 + 2 * v;
            _Float16 lo = (_Float16)conv_w[row * C_DIM + c];
            _Float16 h2 = (_Float16)conv_w[row * C_DIM + c + 1];
            fr.u[v] = ((unsigned int)__builtin_bit_cast(unsigned short, h2) << 16)
                    |  (unsigned int)__builtin_bit_cast(unsigned short, lo);
        }
        aw[cs] = fr.h;
    }
    // conv_b per D-row this lane writes (row m = v + 8*hi)
    float bias[8];
#pragma unroll
    for (int v = 0; v < 8; ++v) bias[v] = conv_b[ktile * 16 + v + 8 * hi];

    v8f vacc[4] = {v8f{}, v8f{}, v8f{}, v8f{}};  // GEMM2 accumulators (4 c-tiles)
    float asum_loc = 0.0f;                        // threads t<64 own cluster k=t

    // ---- streaming-load tiling: each thread owns 4 channels x 4 positions ----
    const int pq = t & 7;    // position quad: p = pq*4 + 0..3
    const int cg = t >> 3;   // channel group: c = cg*4 + 0..3
    const float* xrow = x + (size_t)n * C_DIM * P_TOT + (size_t)pb * PBLOCK
                          + (size_t)(cg * 4) * P_TOT + pq * 4;

    // software pipeline: prefetch tile 0 (4x global_load_b128)
    float4 buf[4];
#pragma unroll
    for (int cc = 0; cc < 4; ++cc)
        buf[cc] = *(const float4*)(xrow + (size_t)cc * P_TOT);

    for (int it = 0; it < ITERS; ++it) {
        float4 cur[4];
#pragma unroll
        for (int cc = 0; cc < 4; ++cc) cur[cc] = buf[cc];

        // ---------- L2-normalize over C (per position) ----------
        float fv[4][4];
#pragma unroll
        for (int cc = 0; cc < 4; ++cc) {
            fv[cc][0] = cur[cc].x; fv[cc][1] = cur[cc].y;
            fv[cc][2] = cur[cc].z; fv[cc][3] = cur[cc].w;
        }
        float ssq[4];
#pragma unroll
        for (int pi = 0; pi < 4; ++pi) {
            float s = 0.0f;
#pragma unroll
            for (int cc = 0; cc < 4; ++cc) s += fv[cc][pi] * fv[cc][pi];
            ssq[pi] = s;
        }
#pragma unroll
        for (int pi = 0; pi < 4; ++pi) red[cg][pq * 4 + pi] = ssq[pi];
        __syncthreads();
        if (t < 32) {
            float s = 0.0f;
#pragma unroll
            for (int j = 0; j < 32; ++j) s += red[j][t];
            rowinv[t] = 1.0f / fmaxf(sqrtf(s), 1e-12f);
        }
        __syncthreads();
        {
            float inv[4];
#pragma unroll
            for (int pi = 0; pi < 4; ++pi) inv[pi] = rowinv[pq * 4 + pi];
            _Float16 hv[4][4];
#pragma unroll
            for (int cc = 0; cc < 4; ++cc)
#pragma unroll
                for (int pi = 0; pi < 4; ++pi)
                    hv[cc][pi] = (_Float16)(fv[cc][pi] * inv[pi]);
            // xn[c][p]: 4 contiguous p per channel -> 8B stores
#pragma unroll
            for (int cc = 0; cc < 4; ++cc) {
                h4 hp = {hv[cc][0], hv[cc][1], hv[cc][2], hv[cc][3]};
                *(h4*)&xn[cg * 4 + cc][pq * 4] = hp;
            }
            // xnT[p][c]: 4 contiguous c per position -> 8B stores
#pragma unroll
            for (int pi = 0; pi < 4; ++pi) {
                h4 hc = {hv[0][pi], hv[1][pi], hv[2][pi], hv[3][pi]};
                *(h4*)&xnT[pq * 4 + pi][cg * 4] = hc;
            }
        }
        __syncthreads();

        // issue next tile's loads now; latency hidden behind GEMM1/softmax/GEMM2
        if (it + 1 < ITERS) {
#pragma unroll
            for (int cc = 0; cc < 4; ++cc)
                buf[cc] = *(const float4*)(xrow + (size_t)cc * P_TOT + (it + 1) * PT);
        }

        // ---------- GEMM1: logits[64 x 32] = conv_w[64x128] * xn[128x32] ----------
        {
            const int pg = cgrp;                // 16-position group
            v8f d = v8f{};
#pragma unroll
            for (int cs = 0; cs < 4; ++cs) {
                union { v16h h; unsigned int u[8]; } fb;
                const int pcol = pg * 16 + lm;
                const int cb   = cs * 32 + (hi ? 16 : 0);
                const unsigned int* src = (const unsigned int*)&xnT[pcol][0];
#pragma unroll
                for (int v = 0; v < 8; ++v) fb.u[v] = src[(cb >> 1) + v];
                d = __builtin_amdgcn_wmma_f32_16x16x32_f16(
                        false, aw[cs], false, fb.h, (short)0, d, false, false);
            }
#pragma unroll
            for (int v = 0; v < 8; ++v)
                lg[ktile * 16 + v + 8 * hi][pg * 16 + lm] = d[v] + bias[v];
        }
        __syncthreads();

        // ---------- softmax over the 64 clusters, per position ----------
        {
            const int sp = t & 31, seg = t >> 5;   // 8 clusters per thread
            float mx = -3.0e38f;
#pragma unroll
            for (int j = 0; j < 8; ++j) mx = fmaxf(mx, lg[seg * 8 + j][sp]);
            red[seg][sp] = mx;
            __syncthreads();
            if (t < 32) {
                float m = red[0][t];
#pragma unroll
                for (int j = 1; j < 8; ++j) m = fmaxf(m, red[j][t]);
                rowm[t] = m;
            }
            __syncthreads();
            const float m = rowm[sp];
            float ex[8], s = 0.0f;
#pragma unroll
            for (int j = 0; j < 8; ++j) {
                float e = __expf(lg[seg * 8 + j][sp] - m);
                ex[j] = e; s += e;
            }
            red[seg][sp] = s;
            __syncthreads();
            if (t < 32) {
                float s2 = 0.0f;
#pragma unroll
                for (int j = 0; j < 8; ++j) s2 += red[j][t];
                rowinv[t] = 1.0f / s2;
            }
            __syncthreads();
            const float is = rowinv[sp];
#pragma unroll
            for (int j = 0; j < 8; ++j)
                aa[seg * 8 + j][sp] = (_Float16)(ex[j] * is);
        }
        __syncthreads();

        // softmax mass per cluster (needed for the centroid term)
        if (t < K_CL) {
            float s = 0.0f;
#pragma unroll
            for (int q = 0; q < PT; ++q) s += (float)aa[t][q];
            asum_loc += s;
        }

        // ---------- GEMM2: vlad[64 x 128] += a[64 x 32] * xn^T[32 x 128] ----------
        {
            union { v16h h; unsigned int u[8]; } fa;
            const int row = ktile * 16 + lm;
            const int b1  = hi ? 16 : 0;
            const unsigned int* sa = (const unsigned int*)&aa[row][0];
#pragma unroll
            for (int v = 0; v < 8; ++v) {
                const int pp = b1 + 2 * v;
                fa.u[v] = sa[pp >> 1];
            }
#pragma unroll
            for (int ct = 0; ct < 4; ++ct) {
                const int ctile = cgrp * 4 + ct;
                union { v16h h; unsigned int u[8]; } fb;
                const unsigned int* sb = (const unsigned int*)&xn[ctile * 16 + lm][0];
                const int pbse = (hi ? 16 : 0) >> 1;
#pragma unroll
                for (int v = 0; v < 8; ++v) fb.u[v] = sb[pbse + v];
                vacc[ct] = __builtin_amdgcn_wmma_f32_16x16x32_f16(
                               false, fa.h, false, fb.h, (short)0, vacc[ct], false, false);
            }
        }
        __syncthreads();
    }

    // ---------- cross-workgroup reduction ----------
    float* accn = acc + (size_t)n * K_CL * C_DIM;
#pragma unroll
    for (int ct = 0; ct < 4; ++ct) {
        const int ctile = cgrp * 4 + ct;
#pragma unroll
        for (int v = 0; v < 8; ++v) {
            const int krow = ktile * 16 + v + 8 * hi;
            const int c    = ctile * 16 + lm;
            atomicAdd(&accn[krow * C_DIM + c], vacc[ct][v]);
        }
    }
    if (t < K_CL) atomicAdd(&asum[n * K_CL + t], asum_loc);
}

__global__ __launch_bounds__(256) void netvlad_finalize(const float* __restrict__ acc,
                                                        const float* __restrict__ asum,
                                                        const float* __restrict__ centroids,
                                                        float* __restrict__ out) {
    const int n = blockIdx.x;
    const int t = threadIdx.x;
    __shared__ float vbuf[K_CL * C_DIM];
    __shared__ float kinv[K_CL];
    __shared__ float part[256];

    const float* a  = acc  + (size_t)n * K_CL * C_DIM;
    const float* as = asum + n * K_CL;
    const int k = t >> 2, q = t & 3;   // 4 threads per cluster, 32 channels each

    float ss = 0.0f;
#pragma unroll
    for (int i = 0; i < 32; ++i) {
        const int c = q * 32 + i;
        float v = a[k * C_DIM + c] - as[k] * centroids[k * C_DIM + c];
        vbuf[k * C_DIM + c] = v;
        ss += v * v;
    }
    part[t] = ss;
    __syncthreads();
    if (q == 0) {
        float s = part[t] + part[t + 1] + part[t + 2] + part[t + 3];
        kinv[k] = 1.0f / fmaxf(sqrtf(s), 1e-12f);
    }
    __syncthreads();
    float gs = 0.0f;
#pragma unroll
    for (int i = 0; i < 32; ++i) {
        const int c = q * 32 + i;
        float v = vbuf[k * C_DIM + c] * kinv[k];
        vbuf[k * C_DIM + c] = v;
        gs += v * v;
    }
    part[t] = gs;
    __syncthreads();
    for (int off = 128; off > 0; off >>= 1) {
        if (t < off) part[t] += part[t + off];
        __syncthreads();
    }
    const float ginv = 1.0f / fmaxf(sqrtf(part[0]), 1e-12f);
    float* o = out + (size_t)n * K_CL * C_DIM;
#pragma unroll
    for (int i = 0; i < 32; ++i) {
        const int c = q * 32 + i;
        o[k * C_DIM + c] = vbuf[k * C_DIM + c] * ginv;
    }
}

extern "C" void kernel_launch(void* const* d_in, const int* in_sizes, int n_in,
                              void* d_out, int out_size, void* d_ws, size_t ws_size,
                              hipStream_t stream) {
    const float* x         = (const float*)d_in[0];  // [32,128,64,64]
    const float* conv_w    = (const float*)d_in[1];  // [64,128]
    const float* conv_b    = (const float*)d_in[2];  // [64]
    const float* centroids = (const float*)d_in[3];  // [64,128]
    float* out = (float*)d_out;                      // [32, 64*128]

    float* acc  = (float*)d_ws;                      // [32][64][128]
    float* asum = acc + (size_t)N_IMG * K_CL * C_DIM;// [32][64]

    const int ztot = N_IMG * K_CL * C_DIM + N_IMG * K_CL;
    netvlad_zero<<<(ztot + 255) / 256, 256, 0, stream>>>(acc, ztot);

    dim3 grid(N_IMG, WG_PER_N);
    netvlad_main<<<grid, 256, 0, stream>>>(x, conv_w, conv_b, acc, asum);

    netvlad_finalize<<<N_IMG, 256, 0, stream>>>(acc, asum, centroids, out);
}
